// Transformer_46978352284027
// MI455X (gfx1250) — matrix operand order repaired and checked by reading, not compiled, not run on hardware
//
#include <hip/hip_runtime.h>
#include <math.h>

#define D_MODEL 768
#define N_HEADS 12
#define DK      64
#define D_FF    3072
#define VOCABSZ 32000
#define TSEQ    2048
#define BATCH   2
#define NLAYERS 6
#define MROWS   (BATCH*TSEQ)   // 4096

typedef _Float16 h16;
typedef __attribute__((ext_vector_type(16))) _Float16 v16h;
typedef __attribute__((ext_vector_type(8)))  float    v8f;

// 16-bit A-matrix (16x32) layout: pair i (elements j=2i,2i+1) sits at
// k = 16*(i>>2) + 2*(i&3) + 8*halfLane   (ISA 7.12.2)
static __device__ __forceinline__ int a_k_idx(int i, int hlf) {
  return 16 * (i >> 2) + 2 * (i & 3) + 8 * hlf;
}

union FragU { v16h v; uint32_t u[8]; };

// ---------------------------------------------------------------------------
// Generic f16 WMMA GEMM: C(MxN) = A(MxK) * B(KxN)  [+bias][+relu]
// Block = 256 threads (8 waves). Block tile 128x128, K-step 32,
// double-buffered LDS. Wave (wm,wn) computes rows wm*32..+32 (2 A-frags),
// cols wn*64..+64 (4 B-frags) -> 8 wmma per K-step.
// ---------------------------------------------------------------------------
__global__ __launch_bounds__(256)
void gemm_wmma_f16(const h16* __restrict__ A, const h16* __restrict__ B,
                   const float* __restrict__ bias,
                   float* __restrict__ Cf, h16* __restrict__ Ch,
                   int N, int K, int relu)
{
  __shared__ __align__(32) h16 As[2][128][40];    // 32 cols + 8 pad
  __shared__ __align__(32) h16 Bs[2][32][144];    // 128 cols + 16 pad (32B rows)

  const int t    = threadIdx.x;
  const int wave = t >> 5, lane = t & 31;
  const int wm   = wave >> 1, wn = wave & 1;
  const int hlf  = lane >> 4, ml = lane & 15;
  const int m0   = blockIdx.y * 128;
  const int n0   = blockIdx.x * 128;

  // global->LDS staging: 16 halfs (32B) per thread for each tile
  const int arow = t >> 1, aseg = t & 1;   // A tile: 128 rows x 32 halfs
  const int brow = t >> 3, bseg = t & 7;   // B tile: 32 rows x 128 halfs
  const h16* Ag = A + (size_t)(m0 + arow) * K + aseg * 16;
  const h16* Bg = B + (size_t)brow * N + n0 + bseg * 16;

  v8f acc[2][4];
#pragma unroll
  for (int s2 = 0; s2 < 2; ++s2)
#pragma unroll
    for (int i = 0; i < 4; ++i)
#pragma unroll
      for (int j = 0; j < 8; ++j) acc[s2][i][j] = 0.0f;

  const int nsteps = K / 32;

  // prologue: stage step 0
  float4 ga0 = ((const float4*)Ag)[0], ga1 = ((const float4*)Ag)[1];
  float4 gb0 = ((const float4*)Bg)[0], gb1 = ((const float4*)Bg)[1];
  {
    float4* da = (float4*)(&As[0][arow][aseg * 16]); da[0] = ga0; da[1] = ga1;
    float4* db = (float4*)(&Bs[0][brow][bseg * 16]); db[0] = gb0; db[1] = gb1;
  }
  __syncthreads();

  for (int s = 0; s < nsteps; ++s) {
    const int cur = s & 1;
    if (s + 1 < nsteps) {                 // fetch next K-slab while computing
      const h16* An = Ag + (size_t)(s + 1) * 32;
      const h16* Bn = Bg + (size_t)(s + 1) * 32 * N;
      ga0 = ((const float4*)An)[0]; ga1 = ((const float4*)An)[1];
      gb0 = ((const float4*)Bn)[0]; gb1 = ((const float4*)Bn)[1];
    }
    if (s + 2 < nsteps) {                 // warm L2 for the slab after next
      __builtin_prefetch(Ag + (size_t)(s + 2) * 32, 0, 0);
      __builtin_prefetch(Bg + (size_t)(s + 2) * 32 * N, 0, 0);
    }

    FragU a[2];
#pragma unroll
    for (int sub = 0; sub < 2; ++sub)
#pragma unroll
      for (int i = 0; i < 8; ++i)
        a[sub].u[i] =
            *(const uint32_t*)(&As[cur][wm * 32 + sub * 16 + ml][a_k_idx(i, hlf)]);

#pragma unroll
    for (int nb = 0; nb < 4; ++nb) {
      v16h b;
#pragma unroll
      for (int j = 0; j < 16; ++j)
        b[j] = Bs[cur][16 * (j >> 3) + 8 * hlf + (j & 7)][wn * 64 + nb * 16 + ml];
      acc[0][nb] = __builtin_amdgcn_wmma_f32_16x16x32_f16(
          false, a[0].v, false, b, (short)0, acc[0][nb], false, false);
      acc[1][nb] = __builtin_amdgcn_wmma_f32_16x16x32_f16(
          false, a[1].v, false, b, (short)0, acc[1][nb], false, false);
    }

    if (s + 1 < nsteps) {
      const int nxt = cur ^ 1;
      float4* da = (float4*)(&As[nxt][arow][aseg * 16]); da[0] = ga0; da[1] = ga1;
      float4* db = (float4*)(&Bs[nxt][brow][bseg * 16]); db[0] = gb0; db[1] = gb1;
    }
    __syncthreads();
  }

#pragma unroll
  for (int sub = 0; sub < 2; ++sub)
#pragma unroll
    for (int nb = 0; nb < 4; ++nb) {
      const int col = n0 + wn * 64 + nb * 16 + ml;
      const float bv = bias ? bias[col] : 0.0f;
#pragma unroll
      for (int r = 0; r < 8; ++r) {
        const int row = m0 + wm * 32 + sub * 16 + r + 8 * hlf;  // C/D layout
        float v = acc[sub][nb][r] + bv;
        if (relu) v = fmaxf(v, 0.0f);
        const size_t off = (size_t)row * N + col;
        if (Cf) Cf[off] = v;
        if (Ch) Ch[off] = (h16)v;
      }
    }
}

// ---------------------------------------------------------------------------
// Flash attention: one wave per (batch, head, 16-query tile).
// Streams 32-key blocks: S = Q K^T (2x2 wmma), online softmax, O += P V
// (P staged f32->f16 through LDS to convert C-layout -> A-layout).
// ---------------------------------------------------------------------------
__global__ __launch_bounds__(32)
void flash_attn(const h16* __restrict__ Q, const h16* __restrict__ Kh,
                const h16* __restrict__ V, const float* __restrict__ padmask,
                const int* __restrict__ causal_p, h16* __restrict__ O)
{
  __shared__ __align__(16) h16 Ps[16][40];

  const int lane = threadIdx.x;
  const int hlf = lane >> 4, ml = lane & 15;
  const int b = blockIdx.z, hd = blockIdx.y, qb = blockIdx.x * 16;
  const int causal = *causal_p;
  const float scale = 0.125f;   // 1/sqrt(64)

  // Q A-fragments for both 32-wide dk slices (loaded once)
  FragU aq[2];
  const h16* qrow = Q + (size_t)(b * TSEQ + qb + ml) * D_MODEL + hd * DK;
#pragma unroll
  for (int s = 0; s < 2; ++s)
#pragma unroll
    for (int i = 0; i < 8; ++i)
      aq[s].u[i] = *(const uint32_t*)(qrow + s * 32 + a_k_idx(i, hlf));

  v8f o[4];
#pragma unroll
  for (int i = 0; i < 4; ++i)
#pragma unroll
    for (int j = 0; j < 8; ++j) o[i][j] = 0.0f;
  float mi[8], li[8];
#pragma unroll
  for (int r = 0; r < 8; ++r) { mi[r] = -INFINITY; li[r] = 0.0f; }

  const int kend = causal ? (qb + 16) : TSEQ;
  for (int kb = 0; kb < kend; kb += 32) {
    float sv[2][8];
#pragma unroll
    for (int t2 = 0; t2 < 2; ++t2) {
      v8f s;
#pragma unroll
      for (int j = 0; j < 8; ++j) s[j] = 0.0f;
#pragma unroll
      for (int sl = 0; sl < 2; ++sl) {
        // B-frag of K^T: lane%16 = key col, element j = contraction row (dk)
        const h16* kcol = Kh + (size_t)(b * TSEQ + kb + t2 * 16 + ml) * D_MODEL
                             + hd * DK + sl * 32;
        v16h bk;
#pragma unroll
        for (int j = 0; j < 16; ++j) bk[j] = kcol[16 * (j >> 3) + 8 * hlf + (j & 7)];
        s = __builtin_amdgcn_wmma_f32_16x16x32_f16(
            false, aq[sl].v, false, bk, (short)0, s, false, false);
      }
      const int keyc = kb + t2 * 16 + ml;
      const float pm = padmask[b * TSEQ + keyc];
#pragma unroll
      for (int r = 0; r < 8; ++r) {
        const int qr = qb + r + 8 * hlf;
        float x = s[r] * scale + pm;
        if (causal && keyc > qr) x = -INFINITY;
        sv[t2][r] = x;
      }
    }

    // online softmax per row (row lives across one 16-lane group)
#pragma unroll
    for (int r = 0; r < 8; ++r) {
      float tm = fmaxf(sv[0][r], sv[1][r]);
#pragma unroll
      for (int off = 8; off >= 1; off >>= 1) tm = fmaxf(tm, __shfl_xor(tm, off, 32));
      const float mnew = fmaxf(mi[r], tm);
      const float alpha = (mi[r] == -INFINITY) ? 0.0f : __expf(mi[r] - mnew);
      const float p0 = (sv[0][r] == -INFINITY) ? 0.0f : __expf(sv[0][r] - mnew);
      const float p1 = (sv[1][r] == -INFINITY) ? 0.0f : __expf(sv[1][r] - mnew);
      float rs = p0 + p1;
#pragma unroll
      for (int off = 8; off >= 1; off >>= 1) rs += __shfl_xor(rs, off, 32);
      li[r] = li[r] * alpha + rs;
      mi[r] = mnew;
#pragma unroll
      for (int dt = 0; dt < 4; ++dt) o[dt][r] *= alpha;
      Ps[r + 8 * hlf][ml]      = (h16)p0;
      Ps[r + 8 * hlf][16 + ml] = (h16)p1;
    }
    __syncthreads();

    FragU ap;
#pragma unroll
    for (int i = 0; i < 8; ++i)
      ap.u[i] = *(const uint32_t*)(&Ps[ml][a_k_idx(i, hlf)]);

    const h16* vrow = V + (size_t)(b * TSEQ + kb) * D_MODEL + hd * DK;
#pragma unroll
    for (int dt = 0; dt < 4; ++dt) {
      v16h bv;
#pragma unroll
      for (int j = 0; j < 16; ++j)
        bv[j] = vrow[(size_t)(16 * (j >> 3) + 8 * hlf + (j & 7)) * D_MODEL + dt * 16 + ml];
      o[dt] = __builtin_amdgcn_wmma_f32_16x16x32_f16(
          false, ap.v, false, bv, (short)0, o[dt], false, false);
    }
    __syncthreads();
  }

#pragma unroll
  for (int r = 0; r < 8; ++r) {
    const float inv = (li[r] > 0.0f) ? 1.0f / li[r] : 0.0f;
    h16* orow = O + (size_t)(b * TSEQ + qb + r + 8 * hlf) * D_MODEL + hd * DK;
#pragma unroll
    for (int dt = 0; dt < 4; ++dt)
      orow[dt * 16 + ml] = (h16)(o[dt][r] * inv);
  }
}

// ---------------------------------------------------------------------------
// Fused residual-add + LayerNorm; writes f32 stream + f16 copy.
// One block (256 threads) per row of 768.
// ---------------------------------------------------------------------------
__global__ __launch_bounds__(256)
void add_ln_k(float* __restrict__ x, const float* __restrict__ y,
              const float* __restrict__ g, const float* __restrict__ beta,
              h16* __restrict__ xh)
{
  __shared__ float red[8];
  const int row = blockIdx.x;
  const int t = threadIdx.x;
  float v[3];
  float s = 0.0f;
#pragma unroll
  for (int i = 0; i < 3; ++i) {
    const int c = t + i * 256;
    float val = x[(size_t)row * D_MODEL + c];
    if (y) val += y[(size_t)row * D_MODEL + c];
    v[i] = val; s += val;
  }
#pragma unroll
  for (int off = 16; off >= 1; off >>= 1) s += __shfl_xor(s, off, 32);
  if ((t & 31) == 0) red[t >> 5] = s;
  __syncthreads();
  float tot = 0.0f;
#pragma unroll
  for (int i = 0; i < 8; ++i) tot += red[i];
  const float mu = tot * (1.0f / D_MODEL);

  float sq = 0.0f;
#pragma unroll
  for (int i = 0; i < 3; ++i) { const float d = v[i] - mu; sq += d * d; }
#pragma unroll
  for (int off = 16; off >= 1; off >>= 1) sq += __shfl_xor(sq, off, 32);
  __syncthreads();
  if ((t & 31) == 0) red[t >> 5] = sq;
  __syncthreads();
  float vt = 0.0f;
#pragma unroll
  for (int i = 0; i < 8; ++i) vt += red[i];
  const float rstd = rsqrtf(vt * (1.0f / D_MODEL) + 1e-5f);

#pragma unroll
  for (int i = 0; i < 3; ++i) {
    const int c = t + i * 256;
    const float out = g[c] * (v[i] - mu) * rstd + beta[c];
    x[(size_t)row * D_MODEL + c] = out;
    xh[(size_t)row * D_MODEL + c] = (h16)out;
  }
}

// ---------------------------------------------------------------------------
// Small helper kernels
// ---------------------------------------------------------------------------
__global__ void conv_f16_k(const float* __restrict__ in, h16* __restrict__ out, size_t n) {
  size_t i = (size_t)blockIdx.x * blockDim.x + threadIdx.x;
  const size_t stride = (size_t)gridDim.x * blockDim.x;
  for (; i < n; i += stride) out[i] = (h16)in[i];
}

__global__ void transpose_embed_k(const float* __restrict__ e, h16* __restrict__ et) {
  const size_t n = (size_t)VOCABSZ * D_MODEL;
  size_t i = (size_t)blockIdx.x * blockDim.x + threadIdx.x;
  const size_t stride = (size_t)gridDim.x * blockDim.x;
  for (; i < n; i += stride) {
    const size_t k = i / VOCABSZ, v = i % VOCABSZ;
    et[i] = (h16)e[v * D_MODEL + k];      // et[k][v] = embed[v][k]
  }
}

__global__ void padmask_k(const int* __restrict__ tokens, float* __restrict__ pm) {
  const int i = blockIdx.x * blockDim.x + threadIdx.x;
  if (i < MROWS) pm[i] = (tokens[i] == 0) ? -INFINITY : 0.0f;
}

__global__ __launch_bounds__(256)
void embed_k(const int* __restrict__ tokens, const float* __restrict__ embed,
             const float* __restrict__ pe, float* __restrict__ x, h16* __restrict__ xh)
{
  const int row = blockIdx.x;        // 0..4095 = b*T + t
  const int t = threadIdx.x;
  const int tt = row % TSEQ;
  const int tok = tokens[row];
  const float sc = 27.712812921102035f;   // sqrt(768)
#pragma unroll
  for (int i = 0; i < 3; ++i) {
    const int c = t + i * 256;
    const float v = embed[(size_t)tok * D_MODEL + c] * sc + pe[(size_t)tt * D_MODEL + c];
    x[(size_t)row * D_MODEL + c] = v;
    xh[(size_t)row * D_MODEL + c] = (h16)v;
  }
}

// ---------------------------------------------------------------------------
// Host orchestration
// ---------------------------------------------------------------------------
extern "C" void kernel_launch(void* const* d_in, const int* in_sizes, int n_in,
                              void* d_out, int out_size, void* d_ws, size_t ws_size,
                              hipStream_t stream) {
  (void)in_sizes; (void)n_in; (void)out_size; (void)ws_size;
  const int*   tokens = (const int*)d_in[0];
  const int*   causal = (const int*)d_in[1];
  const float* embed  = (const float*)d_in[2];
  const float* pe     = (const float*)d_in[3];
  const float* Wq     = (const float*)d_in[4];
  const float* Wk     = (const float*)d_in[5];
  const float* Wv     = (const float*)d_in[6];
  const float* Wo     = (const float*)d_in[7];
  const float* W1     = (const float*)d_in[8];
  const float* b1     = (const float*)d_in[9];
  const float* W2     = (const float*)d_in[10];
  const float* b2     = (const float*)d_in[11];
  const float* g1     = (const float*)d_in[12];
  const float* be1    = (const float*)d_in[13];
  const float* g2     = (const float*)d_in[14];
  const float* be2    = (const float*)d_in[15];
  const float* gf     = (const float*)d_in[16];
  const float* bf     = (const float*)d_in[17];

  char* p = (char*)d_ws;
  auto alloc = [&](size_t bytes) -> void* {
    void* r = (void*)p;
    p += (bytes + 255) & ~(size_t)255;
    return r;
  };
  const size_t WSML = (size_t)NLAYERS * D_MODEL * D_MODEL;   // 6*768*768
  const size_t WLG  = (size_t)NLAYERS * D_MODEL * D_FF;      // 6*768*3072

  h16* Wq_h = (h16*)alloc(sizeof(h16) * WSML);
  h16* Wk_h = (h16*)alloc(sizeof(h16) * WSML);
  h16* Wv_h = (h16*)alloc(sizeof(h16) * WSML);
  h16* Wo_h = (h16*)alloc(sizeof(h16) * WSML);
  h16* W1_h = (h16*)alloc(sizeof(h16) * WLG);
  h16* W2_h = (h16*)alloc(sizeof(h16) * WLG);
  h16* embT = (h16*)alloc(sizeof(h16) * (size_t)VOCABSZ * D_MODEL);
  float* x   = (float*)alloc(sizeof(float) * (size_t)MROWS * D_MODEL);
  h16*   x_h = (h16*)alloc(sizeof(h16) * (size_t)MROWS * D_MODEL);
  h16*   q_h = (h16*)alloc(sizeof(h16) * (size_t)MROWS * D_MODEL);
  h16*   k_h = (h16*)alloc(sizeof(h16) * (size_t)MROWS * D_MODEL);
  h16*   v_h = (h16*)alloc(sizeof(h16) * (size_t)MROWS * D_MODEL);
  h16*   o_h = (h16*)alloc(sizeof(h16) * (size_t)MROWS * D_MODEL);
  float* tmp = (float*)alloc(sizeof(float) * (size_t)MROWS * D_MODEL);
  h16*   h_h = (h16*)alloc(sizeof(h16) * (size_t)MROWS * D_FF);
  float* pm  = (float*)alloc(sizeof(float) * MROWS);

  const dim3 blk(256);
  conv_f16_k<<<dim3(2048), blk, 0, stream>>>(Wq, Wq_h, WSML);
  conv_f16_k<<<dim3(2048), blk, 0, stream>>>(Wk, Wk_h, WSML);
  conv_f16_k<<<dim3(2048), blk, 0, stream>>>(Wv, Wv_h, WSML);
  conv_f16_k<<<dim3(2048), blk, 0, stream>>>(Wo, Wo_h, WSML);
  conv_f16_k<<<dim3(4096), blk, 0, stream>>>(W1, W1_h, WLG);
  conv_f16_k<<<dim3(4096), blk, 0, stream>>>(W2, W2_h, WLG);
  transpose_embed_k<<<dim3(8192), blk, 0, stream>>>(embed, embT);
  padmask_k<<<dim3(16), blk, 0, stream>>>(tokens, pm);
  embed_k<<<dim3(MROWS), blk, 0, stream>>>(tokens, embed, pe, x, x_h);

  const dim3 gD(D_MODEL / 128, MROWS / 128);  // (6,32)
  const dim3 gF(D_FF / 128,    MROWS / 128);  // (24,32)
  const dim3 gV(VOCABSZ / 128, MROWS / 128);  // (250,32)

  for (int l = 0; l < NLAYERS; ++l) {
    const h16* wq = Wq_h + (size_t)l * D_MODEL * D_MODEL;
    const h16* wk = Wk_h + (size_t)l * D_MODEL * D_MODEL;
    const h16* wv = Wv_h + (size_t)l * D_MODEL * D_MODEL;
    const h16* wo = Wo_h + (size_t)l * D_MODEL * D_MODEL;
    const h16* w1 = W1_h + (size_t)l * D_MODEL * D_FF;
    const h16* w2 = W2_h + (size_t)l * D_FF * D_MODEL;

    gemm_wmma_f16<<<gD, blk, 0, stream>>>(x_h, wq, nullptr, nullptr, q_h, D_MODEL, D_MODEL, 0);
    gemm_wmma_f16<<<gD, blk, 0, stream>>>(x_h, wk, nullptr, nullptr, k_h, D_MODEL, D_MODEL, 0);
    gemm_wmma_f16<<<gD, blk, 0, stream>>>(x_h, wv, nullptr, nullptr, v_h, D_MODEL, D_MODEL, 0);

    flash_attn<<<dim3(TSEQ / 16, N_HEADS, BATCH), dim3(32), 0, stream>>>(
        q_h, k_h, v_h, pm, causal, o_h);

    gemm_wmma_f16<<<gD, blk, 0, stream>>>(o_h, wo, nullptr, tmp, nullptr, D_MODEL, D_MODEL, 0);
    add_ln_k<<<dim3(MROWS), blk, 0, stream>>>(x, tmp, g1 + l * D_MODEL, be1 + l * D_MODEL, x_h);

    gemm_wmma_f16<<<gF, blk, 0, stream>>>(x_h, w1, b1 + l * D_FF, nullptr, h_h, D_FF, D_MODEL, 1);
    gemm_wmma_f16<<<gD, blk, 0, stream>>>(h_h, w2, b2 + l * D_MODEL, tmp, nullptr, D_MODEL, D_FF, 0);
    add_ln_k<<<dim3(MROWS), blk, 0, stream>>>(x, tmp, g2 + l * D_MODEL, be2 + l * D_MODEL, x_h);
  }

  add_ln_k<<<dim3(MROWS), blk, 0, stream>>>(x, nullptr, gf, bf, x_h);
  gemm_wmma_f16<<<gV, blk, 0, stream>>>(x_h, embT, nullptr, (float*)d_out, nullptr,
                                        VOCABSZ, D_MODEL, 0);
}
